// HiGNN_39127152066812
// MI455X (gfx1250) — matrix operand order repaired and checked by reading, not compiled, not run on hardware
//
#include <hip/hip_runtime.h>
#include <hip/hip_bf16.h>
#include <math.h>

// ---------------------------------------------------------------------------
// MI455X (gfx1250) HiGNN forward.
// Dense GEMMs -> v_wmma_f32_16x16x32_bf16, wave32, block tile 64x64,
// wave tile 16x32 (2 WMMA / K-step). Double-buffered LDS (1 barrier / K-step,
// global prefetch overlapped with WMMA). Fragment-layout LDS staging:
// A row-contiguous + B transposed [n][k] -> all fragment reads are
// ds_load_b128; A and BT-B global loads are b128, predicated by v_cndmask.
// ---------------------------------------------------------------------------

typedef __attribute__((ext_vector_type(16))) __bf16 v16bf;
typedef __attribute__((ext_vector_type(8)))  float  v8f;

constexpr int BM = 64;
constexpr int BN = 64;
constexpr int BK = 32;
constexpr int LDSP = 40;   // padded row stride in ushorts (80B, keeps 16B align)

union V16 { unsigned int u[8]; v16bf v; };

__device__ inline unsigned int pk2(float a, float b) {
    unsigned short lo = __builtin_bit_cast(unsigned short, (__bf16)a);
    unsigned short hi = __builtin_bit_cast(unsigned short, (__bf16)b);
    return (unsigned int)lo | ((unsigned int)hi << 16);
}

// ---------------- generic WMMA GEMM: O = act(A @ opB + Cadd + bias) ---------
// A: [M,K] f32 row-major (optional row gather via rowIdx)
// B: BT ? [N,K] : [K,N] f32 row-major
// ACT: 0 none, 1 relu, 2 sigmoid
template<int ACT, bool BT, bool HASBIAS, bool HASADD, bool GATHER>
__global__ __launch_bounds__(256)
void wmma_gemm(const float* __restrict__ A, const float* __restrict__ B,
               const float* __restrict__ Cadd, const float* __restrict__ bias,
               float* __restrict__ O, int M, int N, int K,
               int lda, int ldb, int ldo, const int* __restrict__ rowIdx)
{
    __shared__ __align__(16) unsigned short AsU[2][BM][LDSP];
    __shared__ __align__(16) unsigned short BsU[2][BN][LDSP];  // transposed [n][k]

    const int m0b  = blockIdx.x * BM;
    const int n0b  = blockIdx.y * BN;
    const int tid  = threadIdx.x;
    const int wave = tid >> 5;          // 0..7
    const int lane = tid & 31;
    const int wm   = (wave & 3) * 16;   // 0,16,32,48
    const int wn   = (wave >> 2) * 32;  // 0,32

    // A staging: thread -> (row, 8-float chunk)
    const int ar  = tid >> 2;           // 0..63
    const int ac8 = (tid & 3) * 8;      // 0,8,16,24
    const int agm = m0b + ar;
    const int arow = (agm < M) ? (GATHER ? rowIdx[agm] : agm) : 0;
    const float* aptr = A + (size_t)arow * lda + ac8;

    // B staging: thread -> (n, 8-k chunk)
    const int bn  = tid & 63;           // 0..63
    const int bk8 = (tid >> 6) * 8;     // 0,8,16,24
    const int gbn = n0b + bn;
    const bool bokn = gbn < N;
    const float* brow = BT ? (B + (size_t)(bokn ? gbn : 0) * ldb) : B;

    const float4 z4 = {0.f, 0.f, 0.f, 0.f};

    // divergence-free staging loads (address clamp + cndmask zero), packed bf16
    auto loadA = [&](int k0) -> uint4 {
        const bool ok0 = (k0 + ac8)     < K;
        const bool ok1 = (k0 + ac8 + 4) < K;
        float4 f0 = *(const float4*)(ok0 ? (aptr + k0)     : (const float*)A);
        float4 f1 = *(const float4*)(ok1 ? (aptr + k0 + 4) : (const float*)A);
        if (!ok0) f0 = z4;
        if (!ok1) f1 = z4;
        uint4 q;
        q.x = pk2(f0.x, f0.y); q.y = pk2(f0.z, f0.w);
        q.z = pk2(f1.x, f1.y); q.w = pk2(f1.z, f1.w);
        return q;
    };
    auto loadB = [&](int k0) -> uint4 {
        float4 f0, f1;
        if (BT) {
            const bool ok0 = bokn && (k0 + bk8)     < K;
            const bool ok1 = bokn && (k0 + bk8 + 4) < K;
            f0 = *(const float4*)(ok0 ? (brow + k0 + bk8)     : (const float*)B);
            f1 = *(const float4*)(ok1 ? (brow + k0 + bk8 + 4) : (const float*)B);
            if (!ok0) f0 = z4;
            if (!ok1) f1 = z4;
        } else {
            float f[8];
            #pragma unroll
            for (int j = 0; j < 8; ++j) {
                int gk = k0 + bk8 + j;
                bool ok = bokn && gk < K;
                const float* p = B + (size_t)(gk < K ? gk : 0) * ldb + (bokn ? gbn : 0);
                float v = *p;
                f[j] = ok ? v : 0.f;
            }
            f0 = make_float4(f[0], f[1], f[2], f[3]);
            f1 = make_float4(f[4], f[5], f[6], f[7]);
        }
        uint4 q;
        q.x = pk2(f0.x, f0.y); q.y = pk2(f0.z, f0.w);
        q.z = pk2(f1.x, f1.y); q.w = pk2(f1.z, f1.w);
        return q;
    };

    v8f acc0 = {}, acc1 = {};

    // prologue: stage tile 0 into buffer 0
    {
        uint4 qa = loadA(0);
        uint4 qb = loadB(0);
        *(uint4*)&AsU[0][ar][ac8] = qa;
        *(uint4*)&BsU[0][bn][bk8] = qb;
    }
    __syncthreads();

    int buf = 0;
    for (int k0 = 0; k0 < K; k0 += BK) {
        const bool hasNext = (k0 + BK) < K;   // block-uniform
        uint4 qa, qb;
        if (hasNext) {        // prefetch next tile while WMMAs run on `buf`
            qa = loadA(k0 + BK);
            qb = loadB(k0 + BK);
        }

        // ---- fragments: 2x b128 (A) + 4x b128 (B), per ISA 7.12.2 layout ----
        const int mrow = wm + (lane & 15);
        const int ka   = (lane < 16) ? 0 : 8;
        uint4 a0 = *(const uint4*)&AsU[buf][mrow][ka];
        uint4 a1 = *(const uint4*)&AsU[buf][mrow][ka + 16];
        V16 av;
        av.u[0] = a0.x; av.u[1] = a0.y; av.u[2] = a0.z; av.u[3] = a0.w;
        av.u[4] = a1.x; av.u[5] = a1.y; av.u[6] = a1.z; av.u[7] = a1.w;

        const int kb  = (lane < 16) ? 0 : 8;
        const int nc0 = wn + (lane & 15);
        const int nc1 = nc0 + 16;
        uint4 b00 = *(const uint4*)&BsU[buf][nc0][kb];
        uint4 b01 = *(const uint4*)&BsU[buf][nc0][kb + 16];
        uint4 b10 = *(const uint4*)&BsU[buf][nc1][kb];
        uint4 b11 = *(const uint4*)&BsU[buf][nc1][kb + 16];
        V16 bv0, bv1;
        bv0.u[0] = b00.x; bv0.u[1] = b00.y; bv0.u[2] = b00.z; bv0.u[3] = b00.w;
        bv0.u[4] = b01.x; bv0.u[5] = b01.y; bv0.u[6] = b01.z; bv0.u[7] = b01.w;
        bv1.u[0] = b10.x; bv1.u[1] = b10.y; bv1.u[2] = b10.z; bv1.u[3] = b10.w;
        bv1.u[4] = b11.x; bv1.u[5] = b11.y; bv1.u[6] = b11.z; bv1.u[7] = b11.w;

        acc0 = __builtin_amdgcn_wmma_f32_16x16x32_bf16(
                   false, av.v, false, bv0.v, (short)0, acc0, false, false);
        acc1 = __builtin_amdgcn_wmma_f32_16x16x32_bf16(
                   false, av.v, false, bv1.v, (short)0, acc1, false, false);

        if (hasNext) {        // commit prefetched tile to the other buffer
            *(uint4*)&AsU[buf ^ 1][ar][ac8] = qa;
            *(uint4*)&BsU[buf ^ 1][bn][bk8] = qb;
            __syncthreads();
        }
        buf ^= 1;
    }

    // ---- epilogue: C/D layout — lane<16 rows M+0..7, lane>=16 rows M+8..15
    const int mbase = m0b + wm + ((lane < 16) ? 0 : 8);
    const int nout0 = n0b + wn + (lane & 15);
    const int nout1 = nout0 + 16;
    const bool interior = (m0b + BM <= M) && (n0b + BN <= N);   // block-uniform

    auto finish = [&](float v, int nout) -> float {
        if (HASBIAS) v += bias[nout];
        if (ACT == 1) v = fmaxf(v, 0.f);
        else if (ACT == 2) v = 1.f / (1.f + __expf(-v));
        return v;
    };

    if (interior) {
        #pragma unroll
        for (int r = 0; r < 8; ++r) {
            const size_t ro = (size_t)(mbase + r) * ldo;
            float v0 = acc0[r], v1 = acc1[r];
            if (HASADD) { v0 += Cadd[ro + nout0]; v1 += Cadd[ro + nout1]; }
            O[ro + nout0] = finish(v0, nout0);
            O[ro + nout1] = finish(v1, nout1);
        }
    } else {
        #pragma unroll
        for (int r = 0; r < 8; ++r) {
            int mo = mbase + r;
            if (mo < M) {
                const size_t ro = (size_t)mo * ldo;
                if (nout0 < N) {
                    float v = acc0[r];
                    if (HASADD) v += Cadd[ro + nout0];
                    O[ro + nout0] = finish(v, nout0);
                }
                if (nout1 < N) {
                    float v = acc1[r];
                    if (HASADD) v += Cadd[ro + nout1];
                    O[ro + nout1] = finish(v, nout1);
                }
            }
        }
    }
}

// ---------------- helpers ---------------------------------------------------
__device__ inline float wred32(float v) {
    #pragma unroll
    for (int off = 16; off > 0; off >>= 1) v += __shfl_xor(v, off, 32);
    return v;
}
__device__ inline float dot4(float4 a, float4 b) {
    return a.x * b.x + a.y * b.y + a.z * b.z + a.w * b.w;
}
__device__ inline void atomicMaxF(float* addr, float val) {
    unsigned int* a = (unsigned int*)addr;
    unsigned int old = __hip_atomic_load(a, __ATOMIC_RELAXED, __HIP_MEMORY_SCOPE_AGENT);
    while (__uint_as_float(old) < val) {
        unsigned int prev = atomicCAS(a, old, __float_as_uint(val));
        if (prev == old) break;
        old = prev;
    }
}

// ---------------- node-level projections for the "block" term ---------------
// p1[n,s] = dot(xn[n], lin_w[s, 0:128])   (x_i part, gathered at dst)
// p3[n,s] = dot(xn[n], lin_w[s, 256:384]) (x_j part, gathered at src)
// C == 128 assumed (one wave per node, 4 floats/lane).
__global__ __launch_bounds__(256)
void node_proj(const float* __restrict__ xn, const float* __restrict__ lwl,
               float* __restrict__ p1, float* __restrict__ p3, int Nn)
{
    int n    = blockIdx.x * 8 + (threadIdx.x >> 5);
    int lane = threadIdx.x & 31;
    if (n >= Nn) return;
    float4 xv  = ((const float4*)(xn + (size_t)n * 128))[lane];
    float4 w10 = ((const float4*)(lwl + 0))[lane];
    float4 w11 = ((const float4*)(lwl + 384))[lane];
    float4 w30 = ((const float4*)(lwl + 256))[lane];
    float4 w31 = ((const float4*)(lwl + 640))[lane];
    float d10 = wred32(dot4(xv, w10));
    float d11 = wred32(dot4(xv, w11));
    float d30 = wred32(dot4(xv, w30));
    float d31 = wred32(dot4(xv, w31));
    if (lane == 0) {
        p1[n * 2 + 0] = d10; p1[n * 2 + 1] = d11;
        p3[n * 2 + 0] = d30; p3[n * 2 + 1] = d31;
    }
}

// ---------------- fused edge kernel -----------------------------------------
// per edge: score_s = dot(t_s, x_j); block_s = p1[dst]+dot(en,lw2_s)+p3[src]+b
// alpha = tanh(score+block); hh[dst] += max(x_j,en) * alpha[c/64]
__global__ __launch_bounds__(256)
void edge_fused(const float* __restrict__ xn, const float* __restrict__ en,
                const float* __restrict__ t0, const float* __restrict__ t1,
                const float* __restrict__ p1, const float* __restrict__ p3,
                const float* __restrict__ lwl, const float* __restrict__ lbl,
                const int* __restrict__ src, const int* __restrict__ dst,
                float* __restrict__ hh, int E)
{
    int e    = blockIdx.x * 8 + (threadIdx.x >> 5);
    int lane = threadIdx.x & 31;
    if (e >= E) return;
    const int s = src[e], d = dst[e];

    float4 a0 = ((const float4*)(t0 + (size_t)e * 128))[lane];
    float4 a1 = ((const float4*)(t1 + (size_t)e * 128))[lane];
    float4 ev = ((const float4*)(en + (size_t)e * 128))[lane];
    float4 xj = ((const float4*)(xn + (size_t)s * 128))[lane];
    float4 l0 = ((const float4*)(lwl + 128))[lane];        // lin_w[0,128:256]
    float4 l1 = ((const float4*)(lwl + 384 + 128))[lane];  // lin_w[1,128:256]

    float sc0 = wred32(dot4(a0, xj));
    float sc1 = wred32(dot4(a1, xj));
    float pe0 = wred32(dot4(ev, l0));
    float pe1 = wred32(dot4(ev, l1));

    float alpha0 = tanhf(sc0 + p1[d * 2 + 0] + pe0 + p3[s * 2 + 0] + lbl[0]);
    float alpha1 = tanhf(sc1 + p1[d * 2 + 1] + pe1 + p3[s * 2 + 1] + lbl[1]);

    float* out = hh + (size_t)d * 128 + lane * 4;
    float mv[4] = { fmaxf(xj.x, ev.x), fmaxf(xj.y, ev.y),
                    fmaxf(xj.z, ev.z), fmaxf(xj.w, ev.w) };
    #pragma unroll
    for (int i = 0; i < 4; ++i) {
        int c = lane * 4 + i;
        atomicAdd(out + i, mv[i] * (c < 64 ? alpha0 : alpha1));
    }
}

// ---------------- small elementwise kernels ----------------------------------
__global__ void fill_f32(float* p, float v, long long n) {
    long long i = (long long)blockIdx.x * 256 + threadIdx.x;
    if (i < n) p[i] = v;
}
__global__ void relu_ip(float* p, long long n) {
    long long i = (long long)blockIdx.x * 256 + threadIdx.x;
    if (i < n) p[i] = fmaxf(p[i], 0.f);
}
__global__ void fix_mx(float* p, long long n) {
    long long i = (long long)blockIdx.x * 256 + threadIdx.x;
    if (i < n && p[i] < -1e29f) p[i] = 0.f;
}
// gwa = G1+G3, gwb = G2-G3 from gate_w[l] ([C, 3C] row-major)
__global__ void gate_prep(const float* __restrict__ gw, float* gwa, float* gwb, int C) {
    int i = blockIdx.x * 256 + threadIdx.x;
    if (i >= C * C) return;
    int c = i / C, k = i % C;
    const float* row = gw + (size_t)c * 3 * C;
    gwa[i] = row[k] + row[2 * C + k];
    gwb[i] = row[C + k] - row[2 * C + k];
}
__global__ void blend(float* __restrict__ h, const float* __restrict__ hh,
                      const float* __restrict__ beta, long long n) {
    long long i = (long long)blockIdx.x * 256 + threadIdx.x;
    if (i < n) { float b = beta[i]; h[i] = b * h[i] + (1.f - b) * hh[i]; }
}
__global__ void seg_pool(const float* __restrict__ h, const int* __restrict__ batch,
                         float* __restrict__ mx, float* __restrict__ sm, int Nn, int C) {
    long long i = (long long)blockIdx.x * 256 + threadIdx.x;
    if (i >= (long long)Nn * C) return;
    int n = (int)(i / C), c = (int)(i % C);
    int b = batch[n];
    float v = h[i];
    atomicAdd(&sm[(size_t)b * C + c], v);
    atomicMaxF(&mx[(size_t)b * C + c], v);
}
__global__ void pool_add(const float* __restrict__ h, const int* __restrict__ batch,
                         float* __restrict__ mol, int Nn, int C) {
    long long i = (long long)blockIdx.x * 256 + threadIdx.x;
    if (i >= (long long)Nn * C) return;
    int n = (int)(i / C), c = (int)(i % C);
    atomicAdd(&mol[(size_t)batch[n] * C + c], h[i]);
}
__global__ void scale_y(float* __restrict__ h, const float* __restrict__ y,
                        const int* __restrict__ batch, int Nn, int C) {
    long long i = (long long)blockIdx.x * 256 + threadIdx.x;
    if (i >= (long long)Nn * C) return;
    int n = (int)(i / C), c = (int)(i % C);
    h[i] *= y[(size_t)batch[n] * C + c];
}

// ---------------------------------------------------------------------------
extern "C" void kernel_launch(void* const* d_in, const int* in_sizes, int n_in,
                              void* d_out, int out_size, void* d_ws, size_t ws_size,
                              hipStream_t stream)
{
    const float* x         = (const float*)d_in[0];
    const float* edge_attr = (const float*)d_in[1];
    const float* lin_a_w   = (const float*)d_in[2];
    const float* lin_a_b   = (const float*)d_in[3];
    const float* lin_b_w   = (const float*)d_in[4];
    const float* lin_b_b   = (const float*)d_in[5];
    const float* wn        = (const float*)d_in[6];
    const float* we        = (const float*)d_in[7];
    const float* bil_w     = (const float*)d_in[8];
    const float* lin_w     = (const float*)d_in[9];
    const float* lin_bias  = (const float*)d_in[10];
    const float* gate_w    = (const float*)d_in[11];
    const float* gate_b    = (const float*)d_in[12];
    const float* att_w1    = (const float*)d_in[13];
    const float* att_w2    = (const float*)d_in[14];
    const float* out_w     = (const float*)d_in[15];
    const float* out_b     = (const float*)d_in[16];
    const int*   edge_index= (const int*)d_in[17];
    const int*   batch     = (const int*)d_in[18];

    const int Nn  = in_sizes[18];
    const int E   = in_sizes[17] / 2;
    const int IN  = in_sizes[0] / Nn;
    const int ED  = in_sizes[1] / E;
    const int C   = in_sizes[3];
    const int OUT = in_sizes[16];
    const int G   = out_size / OUT;
    const int L   = in_sizes[6] / (C * C);
    const int CR  = in_sizes[13] / (L * C);

    const int* src = edge_index;
    const int* dst = edge_index + E;

    // workspace carve-up (floats)
    float* p    = (float*)d_ws;
    float* h    = p; p += (size_t)Nn * C;
    float* e    = p; p += (size_t)E * C;
    float* xn   = p; p += (size_t)Nn * C;
    float* en   = p; p += (size_t)E * C;
    float* t0   = p; p += (size_t)E * C;
    float* t1   = p; p += (size_t)E * C;
    float* hh   = p; p += (size_t)Nn * C;
    float* tmpb = p; p += (size_t)Nn * C;
    float* beta = p; p += (size_t)Nn * C;
    float* p1   = p; p += (size_t)Nn * 2;
    float* p3   = p; p += (size_t)Nn * 2;
    float* gwa  = p; p += (size_t)C * C;
    float* gwb  = p; p += (size_t)C * C;
    float* mx   = p; p += (size_t)G * C;
    float* sm   = p; p += (size_t)G * C;
    float* z1   = p; p += (size_t)G * CR;
    float* zt   = p; p += (size_t)G * C;
    float* y    = p; p += (size_t)G * C;
    float* mol  = p; p += (size_t)G * C;

    auto gg = [](int M, int N) { return dim3((M + BM - 1) / BM, (N + BN - 1) / BN); };
    auto eb = [](long long n) { return dim3((unsigned)((n + 255) / 256)); };
    const long long NC = (long long)Nn * C, GC = (long long)G * C;

    // h = relu(x @ lin_a_w.T + b) ; e = relu(edge_attr @ lin_b_w.T + b)
    wmma_gemm<1,true,true,false,false><<<gg(Nn,C),256,0,stream>>>(x, lin_a_w, nullptr, lin_a_b, h, Nn, C, IN, IN, IN, C, nullptr);
    wmma_gemm<1,true,true,false,false><<<gg(E,C),256,0,stream>>>(edge_attr, lin_b_w, nullptr, lin_b_b, e, E, C, ED, ED, ED, C, nullptr);

    for (int l = 0; l < L; ++l) {
        const float* wnl = wn     + (size_t)l * C * C;
        const float* wel = we     + (size_t)l * C * C;
        const float* bw0 = bil_w  + (size_t)(l * 2 + 0) * C * C;
        const float* bw1 = bil_w  + (size_t)(l * 2 + 1) * C * C;
        const float* lwl = lin_w  + (size_t)l * 2 * 3 * C;
        const float* lbl = lin_bias + (size_t)l * 2;
        const float* gwl = gate_w + (size_t)l * C * 3 * C;
        const float* gbl = gate_b + (size_t)l * C;
        const float* w1l = att_w1 + (size_t)l * CR * C;
        const float* w2l = att_w2 + (size_t)l * C * CR;

        // node/edge transforms
        wmma_gemm<0,false,false,false,false><<<gg(Nn,C),256,0,stream>>>(h, wnl, nullptr, nullptr, xn, Nn, C, C, C, C, C, nullptr);
        wmma_gemm<0,false,false,false,false><<<gg(E,C),256,0,stream>>>(e, wel, nullptr, nullptr, en, E, C, C, C, C, C, nullptr);
        // node-level block projections
        node_proj<<<(Nn + 7) / 8, 256, 0, stream>>>(xn, lwl, p1, p3, Nn);
        // bilinear: t_s = xn[dst] @ bil_w[l,s]  (gather fused into A loader)
        wmma_gemm<0,false,false,false,true><<<gg(E,C),256,0,stream>>>(xn, bw0, nullptr, nullptr, t0, E, C, C, C, C, C, dst);
        wmma_gemm<0,false,false,false,true><<<gg(E,C),256,0,stream>>>(xn, bw1, nullptr, nullptr, t1, E, C, C, C, C, C, dst);
        // fused edge message + scatter-add
        fill_f32<<<eb(NC),256,0,stream>>>(hh, 0.f, NC);
        edge_fused<<<(E + 7) / 8, 256, 0, stream>>>(xn, en, t0, t1, p1, p3, lwl, lbl, src, dst, hh, E);
        relu_ip<<<eb(NC),256,0,stream>>>(hh, NC);
        // gated residual: beta = sigmoid(h@(G1+G3).T + hh@(G2-G3).T + gb)
        gate_prep<<<eb((long long)C * C),256,0,stream>>>(gwl, gwa, gwb, C);
        wmma_gemm<0,true,false,false,false><<<gg(Nn,C),256,0,stream>>>(h, gwa, nullptr, nullptr, tmpb, Nn, C, C, C, C, C, nullptr);
        wmma_gemm<2,true,true,true,false><<<gg(Nn,C),256,0,stream>>>(hh, gwb, tmpb, gbl, beta, Nn, C, C, C, C, C, nullptr);
        blend<<<eb(NC),256,0,stream>>>(h, hh, beta, NC);
        // feature attention (SE block over graphs)
        fill_f32<<<eb(GC),256,0,stream>>>(mx, -1e30f, GC);
        fill_f32<<<eb(GC),256,0,stream>>>(sm, 0.f, GC);
        seg_pool<<<eb(NC),256,0,stream>>>(h, batch, mx, sm, Nn, C);
        fix_mx<<<eb(GC),256,0,stream>>>(mx, GC);
        wmma_gemm<1,true,false,false,false><<<gg(G,CR),256,0,stream>>>(mx, w1l, nullptr, nullptr, z1, G, CR, C, C, C, CR, nullptr);
        wmma_gemm<0,true,false,false,false><<<gg(G,C),256,0,stream>>>(z1, w2l, nullptr, nullptr, zt, G, C, CR, CR, CR, C, nullptr);
        wmma_gemm<1,true,false,false,false><<<gg(G,CR),256,0,stream>>>(sm, w1l, nullptr, nullptr, z1, G, CR, C, C, C, CR, nullptr);
        wmma_gemm<2,true,false,true,false><<<gg(G,C),256,0,stream>>>(z1, w2l, zt, nullptr, y, G, C, CR, CR, CR, C, nullptr);
        scale_y<<<eb(NC),256,0,stream>>>(h, y, batch, Nn, C);
    }

    // readout: mol = relu(segment_sum(h)); out = mol @ out_w.T + out_b
    fill_f32<<<eb(GC),256,0,stream>>>(mol, 0.f, GC);
    pool_add<<<eb(NC),256,0,stream>>>(h, batch, mol, Nn, C);
    relu_ip<<<eb(GC),256,0,stream>>>(mol, GC);
    wmma_gemm<0,true,true,false,false><<<gg(G,OUT),256,0,stream>>>(mol, out_w, nullptr, out_b, (float*)d_out, G, OUT, C, C, C, OUT, nullptr);
}
